// FineGrainedOpLSTMNet_9680856285275
// MI455X (gfx1250) — matrix operand order repaired and checked by reading, not compiled, not run on hardware
//
#include <hip/hip_runtime.h>
#include <hip/hip_bf16.h>
#include <stddef.h>

// ---------------------------------------------------------------------------
// 2-layer LSTM for MI455X (gfx1250, wave32, WMMA).
//   B=64, T=512, I=H=256, 4H=1024.
// Phase A (per layer): xz = x @ W + b   -- parallel GEMM, bf16 WMMA, f32 acc,
//   2-deep register double-buffering of fragments.
// Phase B (per layer): sequential scan; 4 independent blocks (16 batch rows
//   each, no inter-block sync needed), 16 waves/block, h shared via
//   double-buffered LDS (1 barrier/step), U fragments software-pipelined,
//   next-step xz register-prefetched across the K loop.
// ---------------------------------------------------------------------------

#define BATCH 64
#define TSTEPS 512
#define HDIM 256
#define GDIM 1024          // 4*H
#define MROWS (BATCH * TSTEPS)   // 32768
#define LDSTRIDE 264       // bf16 elems per LDS h row (16B aligned, padded)

typedef __attribute__((ext_vector_type(16))) __bf16 v16bf;
typedef __attribute__((ext_vector_type(8)))  float  v8f;

union F16x16 { v16bf v; uint4 q[2]; };

__device__ __forceinline__ float sigmoidf_(float x) {
    return 1.0f / (1.0f + __expf(-x));
}

// B fragment: 32 contiguous bytes per lane (pre-transposed weight matrix).
__device__ __forceinline__ void load_bfrag(const __bf16* p, F16x16& f) {
    f.q[0] = ((const uint4*)p)[0];
    f.q[1] = ((const uint4*)p)[1];
}
// A fragment: 32 contiguous bytes per lane.
__device__ __forceinline__ void load_afrag(const __bf16* p, F16x16& f) {
    f.q[0] = *(const uint4*)(p);
    f.q[1] = *(const uint4*)(p + 8);
}

// ---------------------------------------------------------------------------
// Converters (run once per call; tiny vs. 134MB xz traffic)
// ---------------------------------------------------------------------------
__global__ void f32_to_bf16_kernel(const float* __restrict__ in,
                                   __bf16* __restrict__ out, int n) {
    int i = blockIdx.x * blockDim.x + threadIdx.x;
    if (i < n) out[i] = (__bf16)in[i];
}

// in: [256][1024] f32 row-major  ->  out: [1024][256] bf16 row-major
__global__ void transpose_to_bf16_kernel(const float* __restrict__ in,
                                         __bf16* __restrict__ out) {
    int tid = blockIdx.x * blockDim.x + threadIdx.x;   // 0 .. 262143
    int n = tid >> 8;          // 0..1023
    int k = tid & 255;         // 0..255
    out[tid] = (__bf16)in[k * GDIM + n];
}

// ---------------------------------------------------------------------------
// Phase A: C[M,1024] = A_bf16[M,256] x Bt_bf16[1024,256]^T + bias (f32 out)
// Block: 256 threads (8 waves). Block tile: 16 rows x 1024 cols.
// Wave w: cols [128w, 128w+128) -> 8 WMMA tiles, K=256 -> 8 k-steps.
// Fragments double-buffered so k-step ks+1's loads overlap ks's WMMAs.
// ---------------------------------------------------------------------------
__global__ void __launch_bounds__(256)
gemm_xz_kernel(const __bf16* __restrict__ A,    // [M][256]
               const __bf16* __restrict__ Bt,   // [1024][256] (pre-transposed)
               const float*  __restrict__ bias, // [1024]
               float* __restrict__ C) {         // [M][1024]
    const int lane = threadIdx.x & 31;
    const int wave = threadIdx.x >> 5;       // 0..7
    const int half = lane >> 4;              // 0/1 (lane group)
    const int l16  = lane & 15;
    const size_t row0 = (size_t)blockIdx.x * 16;
    const int cb = wave * 128;

    v8f acc[8];
#pragma unroll
    for (int tn = 0; tn < 8; ++tn) {
        float bv = bias[cb + tn * 16 + l16];
#pragma unroll
        for (int r = 0; r < 8; ++r) acc[tn][r] = bv;
    }

    const __bf16* arow = A + (row0 + (size_t)l16) * 256 + half * 16;
    const __bf16* brow = Bt + ((size_t)(cb + l16) * 256 + half * 16);

    F16x16 aF[2];
    F16x16 bF[2][8];

    // prologue: k-step 0
    load_afrag(arow, aF[0]);
#pragma unroll
    for (int tn = 0; tn < 8; ++tn)
        load_bfrag(brow + (size_t)tn * 16 * 256, bF[0][tn]);

#pragma unroll
    for (int ks = 0; ks < 8; ++ks) {
        const int cur = ks & 1;
        const int nxt = cur ^ 1;
        if (ks < 7) {
            const int k0 = (ks + 1) * 32;
            load_afrag(arow + k0, aF[nxt]);
#pragma unroll
            for (int tn = 0; tn < 8; ++tn)
                load_bfrag(brow + (size_t)tn * 16 * 256 + k0, bF[nxt][tn]);
        }
#pragma unroll
        for (int tn = 0; tn < 8; ++tn)
            acc[tn] = __builtin_amdgcn_wmma_f32_16x16x32_bf16(
                false, aF[cur].v, false, bF[cur][tn].v, (short)0, acc[tn],
                false, false);
    }

    // C/D layout: VGPR r -> M = r + 8*half, N = l16
#pragma unroll
    for (int tn = 0; tn < 8; ++tn) {
        const int col = cb + tn * 16 + l16;
#pragma unroll
        for (int r = 0; r < 8; ++r)
            C[(row0 + r + half * 8) * GDIM + col] = acc[tn][r];
    }
}

// ---------------------------------------------------------------------------
// Phase B: LSTM scan. grid=4 (batch chunks of 16), block=512 (16 waves).
// Wave w owns h columns [16w,16w+16); computes gate tiles i,f,g,o (z cols
// g*256 + [16w,16w+16)), K=256 over LDS-resident h (bf16, double buffered).
// Gate-major K loop: A frags loaded once per step (LDS), B frags (U, L2-hot)
// pipelined in half-gate chunks; next-step xz prefetched into registers.
// ---------------------------------------------------------------------------
template <bool FINAL>
__global__ void __launch_bounds__(512)
lstm_scan_kernel(const float*  __restrict__ xz,   // [B][T][1024]
                 const __bf16* __restrict__ Ut,   // [1024][256] pre-transposed
                 __bf16* __restrict__ out_bf,     // [B][T][256] (FINAL=false)
                 float*  __restrict__ out_f32) {  // [B][T][256] (FINAL=true)
    __shared__ __align__(16) __bf16 hbuf[2][16 * LDSTRIDE];

    const int lane = threadIdx.x & 31;
    const int wave = threadIdx.x >> 5;    // 0..15
    const int half = lane >> 4;
    const int l16  = lane & 15;
    const int b0   = blockIdx.x * 16;     // batch chunk
    const int hcol = wave * 16 + l16;     // this lane's h / N column

    // zero h(t=-1)
    for (int i = threadIdx.x; i < 2 * 16 * LDSTRIDE; i += blockDim.x)
        (&hbuf[0][0])[i] = (__bf16)0.0f;
    __syncthreads();

    v8f c_state = {0.f, 0.f, 0.f, 0.f, 0.f, 0.f, 0.f, 0.f};

    // Per-lane base for B fragments (time-invariant; L2-resident).
    // Gate g tile column = g*256 + hcol; lane chunk at k0 + 16*half.
    const __bf16* ubase = Ut + ((size_t)hcol * 256 + half * 16);
    // xz element for (gate g, reg r): [(b0 + r + 8*half), t, g*256 + hcol]
    const float* xzbase =
        xz + (size_t)(b0 + half * 8) * TSTEPS * GDIM + hcol;

    // ---- prefetch xz(t=0) into registers ----
    v8f xznext[4];
#pragma unroll
    for (int g = 0; g < 4; ++g)
#pragma unroll
        for (int r = 0; r < 8; ++r)
            xznext[g][r] =
                xzbase[(size_t)r * TSTEPS * GDIM + (size_t)g * 256];

    int cur = 0;
    for (int t = 0; t < TSTEPS; ++t) {
        // --- consume prefetched xz(t), then issue loads for xz(t+1);
        //     the loads stay outstanding across the whole K loop. ---
        v8f acc[4];
#pragma unroll
        for (int g = 0; g < 4; ++g) acc[g] = xznext[g];
        if (t + 1 < TSTEPS) {
            const float* xzt1 = xzbase + (size_t)(t + 1) * GDIM;
#pragma unroll
            for (int g = 0; g < 4; ++g)
#pragma unroll
                for (int r = 0; r < 8; ++r)
                    xznext[g][r] =
                        xzt1[(size_t)r * TSTEPS * GDIM + (size_t)g * 256];
        }

        // --- A fragments: h(t-1) from LDS, loaded once, reused by all gates
        const __bf16* hrow = &hbuf[cur][(size_t)l16 * LDSTRIDE + half * 16];
        F16x16 aF[8];
#pragma unroll
        for (int ks = 0; ks < 8; ++ks)
            load_afrag(hrow + ks * 32, aF[ks]);

        // --- z += h(t-1) @ U, gate-major, B pipelined in half-gate chunks
        //     (chunk c -> gate c>>1, k-range (c&1)*128..+128, 4 wmma). ---
        F16x16 bF[2][4];
#pragma unroll
        for (int j = 0; j < 4; ++j)                      // prologue: chunk 0
            load_bfrag(ubase + (size_t)j * 32, bF[0][j]);
#pragma unroll
        for (int c = 0; c < 8; ++c) {
            const int g   = c >> 1;
            const int hk  = c & 1;
            const int buf = c & 1;       // ping-pong
            if (c < 7) {
                const int gn  = (c + 1) >> 1;
                const int hkn = (c + 1) & 1;
                const __bf16* ub =
                    ubase + (size_t)gn * 256 * 256 + (size_t)hkn * 128;
#pragma unroll
                for (int j = 0; j < 4; ++j)
                    load_bfrag(ub + (size_t)j * 32, bF[buf ^ 1][j]);
            }
#pragma unroll
            for (int j = 0; j < 4; ++j)
                acc[g] = __builtin_amdgcn_wmma_f32_16x16x32_bf16(
                    false, aF[hk * 4 + j].v, false, bF[buf][j].v, (short)0,
                    acc[g], false, false);
        }

        // --- gates (fp32): c = sig(f)*c + sig(i)*tanh(g); h = sig(o)*tanh(c)
        const int nxt = cur ^ 1;
        __bf16* hw = &hbuf[nxt][0];
        float hnew[8];
#pragma unroll
        for (int r = 0; r < 8; ++r) {
            const float ig = sigmoidf_(acc[0][r]);
            const float fg = sigmoidf_(acc[1][r]);
            const float gg = tanhf(acc[2][r]);
            const float og = sigmoidf_(acc[3][r]);
            const float c  = fg * c_state[r] + ig * gg;
            c_state[r] = c;
            hnew[r] = og * tanhf(c);
        }

        // --- publish h(t): LDS (next step's A) + global layer output ---
#pragma unroll
        for (int r = 0; r < 8; ++r) {
            const int row = r + half * 8;                  // batch row in chunk
            const __bf16 hb = (__bf16)hnew[r];
            hw[(size_t)row * LDSTRIDE + hcol] = hb;
            const size_t oidx =
                (((size_t)(b0 + row)) * TSTEPS + t) * HDIM + hcol;
            if (FINAL) out_f32[oidx] = hnew[r];
            else       out_bf[oidx]  = hb;
        }
        __syncthreads();   // all waves done reading buf[cur] / writing buf[nxt]
        cur = nxt;
    }
}

// ---------------------------------------------------------------------------
// Host launcher.
//   inputs: x[B,T,I], W0[I,4H], U0[H,4H], b0[4H], W1[H,4H], U1[H,4H], b1[4H]
//   output: y[B,T,H] f32
// Workspace layout (≈170 MB):
//   xz      32768*1024 f32   = 134,217,728 B   (L2-resident on MI455X, 192MB)
//   xbf     32768* 256 bf16  =  16,777,216 B
//   ybf     32768* 256 bf16  =  16,777,216 B
//   Wt0,Ut0,Wt1,Ut1  each 1024*256 bf16 = 524,288 B
// ---------------------------------------------------------------------------
extern "C" void kernel_launch(void* const* d_in, const int* in_sizes, int n_in,
                              void* d_out, int out_size, void* d_ws,
                              size_t ws_size, hipStream_t stream) {
    const float* x  = (const float*)d_in[0];
    const float* W0 = (const float*)d_in[1];
    const float* U0 = (const float*)d_in[2];
    const float* b0 = (const float*)d_in[3];
    const float* W1 = (const float*)d_in[4];
    const float* U1 = (const float*)d_in[5];
    const float* b1 = (const float*)d_in[6];

    char* ws = (char*)d_ws;
    size_t off = 0;
    float*  xz  = (float*)(ws + off);  off += (size_t)MROWS * GDIM * 4;
    __bf16* xbf = (__bf16*)(ws + off); off += (size_t)MROWS * HDIM * 2;
    __bf16* ybf = (__bf16*)(ws + off); off += (size_t)MROWS * HDIM * 2;
    __bf16* Wt0 = (__bf16*)(ws + off); off += (size_t)GDIM * HDIM * 2;
    __bf16* Ut0 = (__bf16*)(ws + off); off += (size_t)GDIM * HDIM * 2;
    __bf16* Wt1 = (__bf16*)(ws + off); off += (size_t)GDIM * HDIM * 2;
    __bf16* Ut1 = (__bf16*)(ws + off); off += (size_t)GDIM * HDIM * 2;

    const int nx = MROWS * HDIM;   // 8,388,608
    f32_to_bf16_kernel<<<(nx + 255) / 256, 256, 0, stream>>>(x, xbf, nx);
    transpose_to_bf16_kernel<<<1024, 256, 0, stream>>>(W0, Wt0);
    transpose_to_bf16_kernel<<<1024, 256, 0, stream>>>(U0, Ut0);
    transpose_to_bf16_kernel<<<1024, 256, 0, stream>>>(W1, Wt1);
    transpose_to_bf16_kernel<<<1024, 256, 0, stream>>>(U1, Ut1);

    // ----- layer 0 -----
    gemm_xz_kernel<<<MROWS / 16, 256, 0, stream>>>(xbf, Wt0, b0, xz);
    lstm_scan_kernel<false><<<BATCH / 16, 512, 0, stream>>>(
        xz, Ut0, ybf, (float*)nullptr);

    // ----- layer 1 -----
    gemm_xz_kernel<<<MROWS / 16, 256, 0, stream>>>(ybf, Wt1, b1, xz);
    lstm_scan_kernel<true><<<BATCH / 16, 512, 0, stream>>>(
        xz, Ut1, (__bf16*)nullptr, (float*)d_out);
}
